// Co_attention_32040456028462
// MI455X (gfx1250) — matrix operand order repaired and checked by reading, not compiled, not run on hardware
//
#include <hip/hip_runtime.h>
#include <hip/hip_bf16.h>

#define B_  32
#define LH_ 2048
#define LT_ 2048
#define D_  768

typedef __attribute__((ext_vector_type(16))) __bf16 v16bf;
typedef __attribute__((ext_vector_type(8)))  __bf16 v8bf;
typedef __attribute__((ext_vector_type(8)))  float  v8f;
typedef __attribute__((ext_vector_type(4)))  unsigned v4u;
typedef __attribute__((ext_vector_type(8)))  int      v8i;
typedef __attribute__((ext_vector_type(4)))  int      v4i;

// ---------------------------------------------------------------------------
// TDM 2D tile load: global (row-major, 2-byte elems) -> LDS.
// D# packing per CDNA5 ISA (2D: groups 2/3 zero).
// 6-arg builtin on this toolchain: (v4u, v8i, v4i, v4i, v8i, i32 cpol).
// ---------------------------------------------------------------------------
static __device__ __forceinline__ void tdm_load_2d_b16(
        unsigned lds_off, const void* gptr,
        unsigned tile_w, unsigned tile_h,
        unsigned tensor_w, unsigned tensor_h,
        unsigned row_stride) {
    unsigned long long ga = (unsigned long long)gptr;
    v4u g0;
    g0[0] = 1u;                                         // count=1, user descriptor
    g0[1] = lds_off;                                    // lds_addr[31:0]
    g0[2] = (unsigned)(ga & 0xFFFFFFFFull);             // global_addr[31:0]
    g0[3] = (unsigned)((ga >> 32) & 0x1FFFFFFull)       // global_addr[56:32]
          | (2u << 30);                                 // type = 2 ("image")
    v8i g1;
    g1[0] = (int)(1u << 16);                            // data_size=1 -> 2 bytes
    g1[1] = (int)((tensor_w & 0xFFFFu) << 16);          // tensor_dim0[15:0]
    g1[2] = (int)((tensor_w >> 16) |
                  ((tensor_h & 0xFFFFu) << 16));        // tensor_dim1[15:0]
    g1[3] = (int)((tensor_h >> 16) |
                  ((tile_w & 0xFFFFu) << 16));          // tile_dim0
    g1[4] = (int)(tile_h & 0xFFFFu);                    // tile_dim1 (tile_dim2=0 -> 2D)
    g1[5] = (int)row_stride;                            // tensor_dim0_stride[31:0]
    g1[6] = 0;
    g1[7] = 0;
    v4i gz4 = {0, 0, 0, 0};
    v8i gz8 = {0, 0, 0, 0, 0, 0, 0, 0};
    __builtin_amdgcn_tensor_load_to_lds(g0, g1, gz4, gz4, gz8, 0);
}

// A-fragment gather: 16 contiguous k per lane half (k = 16*half + s)
static __device__ __forceinline__ v16bf ldA(const __bf16* p) {
    v8bf lo = *(const v8bf*)p;
    v8bf hi = *(const v8bf*)(p + 8);
    return __builtin_shufflevector(lo, hi,
               0,1,2,3,4,5,6,7,8,9,10,11,12,13,14,15);
}

// ---------------------------------------------------------------------------
// Kernel 0a: Hb = bf16(H)
// ---------------------------------------------------------------------------
__global__ void __launch_bounds__(256) prep_h_kernel(const float* __restrict__ H,
                                                     __bf16* __restrict__ Hb) {
    size_t i = ((size_t)blockIdx.x * 256 + threadIdx.x) * 8;
    float4 f0 = *(const float4*)(H + i);
    float4 f1 = *(const float4*)(H + i + 4);
    v8bf o;
    o[0] = (__bf16)f0.x; o[1] = (__bf16)f0.y; o[2] = (__bf16)f0.z; o[3] = (__bf16)f0.w;
    o[4] = (__bf16)f1.x; o[5] = (__bf16)f1.y; o[6] = (__bf16)f1.z; o[7] = (__bf16)f1.w;
    *(v8bf*)(Hb + i) = o;
}

// ---------------------------------------------------------------------------
// Kernel 0b: Wt[e][d] = bf16(W[d][e])
// ---------------------------------------------------------------------------
__global__ void prep_w_kernel(const float* __restrict__ W, __bf16* __restrict__ Wt) {
    int e = blockIdx.x * 16 + threadIdx.x;
    int d = blockIdx.y * 16 + threadIdx.y;
    Wt[(size_t)e * D_ + d] = (__bf16)W[(size_t)d * D_ + e];
}

// ---------------------------------------------------------------------------
// Kernel 1: z1[b] = H[b] @ W
// 4 waves/block, 64 e-cols; B register-resident; A tiles TDM-DMA'd into
// double-buffered LDS (wave 0 issues; TENSORcnt + barrier publish).
// ---------------------------------------------------------------------------
__global__ void __launch_bounds__(128) gemm1_kernel(const __bf16* __restrict__ Hb,
                                                    const __bf16* __restrict__ Wt,
                                                    __bf16* __restrict__ z1) {
    __shared__ __align__(16) __bf16 sA[2][16][D_];  // 2 x 24 KB
    const int b      = blockIdx.z;
    const int eBlock = blockIdx.x * 64;
    const int tid    = threadIdx.x;
    const int wave = tid >> 5, lane = tid & 31;
    const int n = lane & 15, half = lane >> 4;
    const int ecol = eBlock + wave * 16 + n;

    v16bf Breg[24];
#pragma unroll
    for (int kk = 0; kk < D_ / 32; ++kk)
        Breg[kk] = *(const v16bf*)(Wt + (size_t)ecol * D_ + kk * 32 + half * 16);

    const int mTiles = (LH_ / 16) / 4;
    const int m0Base = blockIdx.y * mTiles * 16;
    const __bf16* hb = Hb + (size_t)b * LH_ * D_;

    if (wave == 0)
        tdm_load_2d_b16((unsigned)(size_t)&sA[0][0][0],
                        hb + (size_t)m0Base * D_, D_, 16, D_, LH_, D_);

    for (int mt = 0; mt < mTiles; ++mt) {
        const int cur = mt & 1;
        if (wave == 0) {
            if (mt + 1 < mTiles) {
                tdm_load_2d_b16((unsigned)(size_t)&sA[1 - cur][0][0],
                                hb + (size_t)(m0Base + (mt + 1) * 16) * D_,
                                D_, 16, D_, LH_, D_);
                __builtin_amdgcn_s_wait_tensorcnt(1);
            } else {
                __builtin_amdgcn_s_wait_tensorcnt(0);
            }
        }
        __syncthreads();                            // tile published to all waves

        const __bf16* abase = &sA[cur][n][half * 16];
        v8f acc = {};
        v16bf a0 = ldA(abase);                      // 1-deep software pipeline
#pragma unroll
        for (int kk = 0; kk < D_ / 32; ++kk) {
            v16bf a1 = a0;
            if (kk + 1 < D_ / 32) a1 = ldA(abase + (kk + 1) * 32);
            acc = __builtin_amdgcn_wmma_f32_16x16x32_bf16(
                      false, a0, false, Breg[kk], (short)0, acc, false, false);
            a0 = a1;
        }
        const int m0 = m0Base + mt * 16;
        __bf16* zp = z1 + ((size_t)b * LH_ + m0 + half * 8) * D_ + ecol;
#pragma unroll
        for (int v = 0; v < 8; ++v)
            zp[(size_t)v * D_] = (__bf16)acc[v];
        __syncthreads();                            // release buffer
    }
}

// ---------------------------------------------------------------------------
// Kernel 2: alpha[b][t] = tanh( max_l (z1[b] @ T[b]^T)[l][t] )
// tanh(max)==max(tanh): the [B,LH,LT] tensor is never materialized.
// ---------------------------------------------------------------------------
__global__ void __launch_bounds__(128) gemm2_kernel(const __bf16* __restrict__ z1,
                                                    const float* __restrict__ T,
                                                    float* __restrict__ alpha) {
    __shared__ __align__(16) __bf16 sA[2][16][D_];  // 2 x 24 KB
    const int b      = blockIdx.y;
    const int tBlock = blockIdx.x * 64;
    const int tid    = threadIdx.x;
    const int wave = tid >> 5, lane = tid & 31;
    const int n = lane & 15, half = lane >> 4;
    const int tcol = tBlock + wave * 16 + n;

    v16bf Breg[24];
#pragma unroll
    for (int kk = 0; kk < D_ / 32; ++kk) {
        const float* tp = T + ((size_t)b * LT_ + tcol) * D_ + kk * 32 + half * 16;
        v16bf bf;
#pragma unroll
        for (int i = 0; i < 16; ++i) bf[i] = (__bf16)tp[i];
        Breg[kk] = bf;
    }

    const __bf16* zb = z1 + (size_t)b * LH_ * D_;

    if (wave == 0)
        tdm_load_2d_b16((unsigned)(size_t)&sA[0][0][0], zb, D_, 16, D_, LH_, D_);

    float rmax = -3.402823466e38f;
    for (int mt = 0; mt < LH_ / 16; ++mt) {
        const int cur = mt & 1;
        if (wave == 0) {
            if (mt + 1 < LH_ / 16) {
                tdm_load_2d_b16((unsigned)(size_t)&sA[1 - cur][0][0],
                                zb + (size_t)(mt + 1) * 16 * D_, D_, 16, D_, LH_, D_);
                __builtin_amdgcn_s_wait_tensorcnt(1);
            } else {
                __builtin_amdgcn_s_wait_tensorcnt(0);
            }
        }
        __syncthreads();

        const __bf16* abase = &sA[cur][n][half * 16];
        v8f acc = {};
        v16bf a0 = ldA(abase);
#pragma unroll
        for (int kk = 0; kk < D_ / 32; ++kk) {
            v16bf a1 = a0;
            if (kk + 1 < D_ / 32) a1 = ldA(abase + (kk + 1) * 32);
            acc = __builtin_amdgcn_wmma_f32_16x16x32_bf16(
                      false, a0, false, Breg[kk], (short)0, acc, false, false);
            a0 = a1;
        }
#pragma unroll
        for (int v = 0; v < 8; ++v) rmax = fmaxf(rmax, acc[v]);
        __syncthreads();
    }
    rmax = fmaxf(rmax, __shfl_xor(rmax, 16, 32));
    if (half == 0)
        alpha[(size_t)b * LT_ + tcol] = tanhf(rmax);
}

// ---------------------------------------------------------------------------
// Kernel 3a/3b: HT[b][d] = sum_t alpha[b][t]*T[b][t][d] (deterministic 2-stage)
// ---------------------------------------------------------------------------
__global__ void __launch_bounds__(256) ht_partial_kernel(const float* __restrict__ T,
                                                         const float* __restrict__ alpha,
                                                         float* __restrict__ partial) {
    __shared__ float sAl[256];
    const int b     = blockIdx.y;
    const int chunk = blockIdx.z;
    const int d     = blockIdx.x * 256 + threadIdx.x;
    const int t0    = chunk * 256;
    sAl[threadIdx.x] = alpha[(size_t)b * LT_ + t0 + threadIdx.x];
    __syncthreads();
    float sum = 0.f;
    const float* tp = T + ((size_t)b * LT_ + t0) * D_ + d;
#pragma unroll 4
    for (int t = 0; t < 256; ++t)
        sum += sAl[t] * tp[(size_t)t * D_];
    partial[((size_t)chunk * B_ + b) * D_ + d] = sum;
}

__global__ void __launch_bounds__(256) ht_reduce_kernel(const float* __restrict__ partial,
                                                        float* __restrict__ out) {
    const int b = blockIdx.y;
    const int d = blockIdx.x * 256 + threadIdx.x;
    float s = 0.f;
#pragma unroll
    for (int c = 0; c < 8; ++c)
        s += partial[((size_t)c * B_ + b) * D_ + d];
    out[(size_t)b * D_ + d] = s;
}

// ---------------------------------------------------------------------------
extern "C" void kernel_launch(void* const* d_in, const int* in_sizes, int n_in,
                              void* d_out, int out_size, void* d_ws, size_t ws_size,
                              hipStream_t stream) {
    const float* H = (const float*)d_in[0];
    const float* T = (const float*)d_in[1];
    const float* W = (const float*)d_in[2];
    float* out = (float*)d_out;

    char* ws = (char*)d_ws;
    const size_t hbBytes = (size_t)B_ * LH_ * D_ * sizeof(__bf16);
    const size_t z1Bytes = (size_t)B_ * LH_ * D_ * sizeof(__bf16);
    const size_t wtBytes = (size_t)D_ * D_ * sizeof(__bf16);
    const size_t alBytes = (size_t)B_ * LT_ * sizeof(float);

    __bf16* Hb    = (__bf16*)ws;
    __bf16* z1    = (__bf16*)(ws + hbBytes);
    __bf16* Wt    = (__bf16*)(ws + hbBytes + z1Bytes);
    float*  alpha = (float*)(ws + hbBytes + z1Bytes + wtBytes);
    float*  parts = (float*)(ws + hbBytes + z1Bytes + wtBytes + alBytes);

    const int hElems = B_ * LH_ * D_;
    prep_h_kernel<<<dim3(hElems / (256 * 8)), 256, 0, stream>>>(H, Hb);
    prep_w_kernel<<<dim3(D_ / 16, D_ / 16), dim3(16, 16), 0, stream>>>(W, Wt);
    gemm1_kernel<<<dim3(D_ / 64, 4, B_), 128, 0, stream>>>(Hb, Wt, z1);
    gemm2_kernel<<<dim3(LT_ / 64, B_), 128, 0, stream>>>(z1, T, alpha);
    ht_partial_kernel<<<dim3(D_ / 256, B_, 8), 256, 0, stream>>>(T, alpha, parts);
    ht_reduce_kernel<<<dim3(D_ / 256, B_), 256, 0, stream>>>(parts, out);
}
